// YOSOAttention_69965017252059
// MI455X (gfx1250) — matrix operand order repaired and checked by reading, not compiled, not run on hardware
//
#include <hip/hip_runtime.h>
#include <stdint.h>

namespace {
constexpr int kB = 2, kH = 12, kS = 2048, kD = 64;
constexpr int kKer = 9;  // conv kernel size, pad 4
}

typedef __attribute__((ext_vector_type(16))) _Float16 v16h;
typedef __attribute__((ext_vector_type(8)))  float    v8f;

union Frag {
  v16h     h;
  uint32_t u[8];
};

static __device__ __forceinline__ v8f wmma_f16(v16h a, v16h b, v8f c) {
  // D = A(16x32 f16) * B(32x16 f16) + C(16x16 f32)
  return __builtin_amdgcn_wmma_f32_16x16x32_f16(false, a, false, b, (short)0, c,
                                                false, false);
}

// Raw-hardware helpers: single v_sqrt_f32 / v_rcp_f32 (~1 ulp), no IEEE
// refinement sequences. Arguments here are clamped well inside normal range.
static __device__ __forceinline__ float fast_sqrt(float x) {
  return __builtin_amdgcn_sqrtf(x);
}
static __device__ __forceinline__ float fast_rcp(float x) {
  return __builtin_amdgcn_rcpf(x);
}

// Fast t = 1 - acos(x)/pi for x in [-1,1].
// Abramowitz-Stegun 4.4.45: acos(|x|) ~= sqrt(1-|x|) * poly(|x|), |err|<=6.8e-5.
// Reflection folded into t:  x>=0: t = 1 - s*p/pi ;  x<0: t = s*p(-x)/pi.
static __device__ __forceinline__ float one_minus_acos_over_pi(float x) {
  float xa = fabsf(x);
  float s  = fast_sqrt(fmaxf(1.0f - xa, 0.0f));
  float p  = fmaf(fmaf(fmaf(-0.0187293f, xa, 0.0742610f), xa, -0.2121144f), xa,
                  1.5707288f);
  float u  = s * p * 0.3183098861837907f;  // acos(|x|)/pi
  return (x >= 0.0f) ? (1.0f - u) : u;
}

// ---------------------------------------------------------------------------
// Prep: L2-normalize Q,K -> f16; convert V -> f16; depthwise conv over seq.
// One wave per (b,h,s) row; each lane handles d = lane and lane+32.
// ---------------------------------------------------------------------------
__global__ __launch_bounds__(256) void yoso_prep(
    const float* __restrict__ Q, const float* __restrict__ K,
    const float* __restrict__ V, const float* __restrict__ mask,
    const float* __restrict__ convw,
    unsigned short* __restrict__ QhU, unsigned short* __restrict__ KhU,
    unsigned short* __restrict__ VhU, float* __restrict__ convV) {
  _Float16* Qh = reinterpret_cast<_Float16*>(QhU);
  _Float16* Kh = reinterpret_cast<_Float16*>(KhU);
  _Float16* Vh = reinterpret_cast<_Float16*>(VhU);

  const int row  = blockIdx.x * 8 + (threadIdx.x >> 5);
  const int lane = threadIdx.x & 31;
  const int s    = row % kS;
  const int bh   = row / kS;
  const int h    = bh % kH;
  const int b    = bh / kH;
  const size_t off = (size_t)row * kD;

  {  // Q normalize -> f16
    float x0 = Q[off + lane], x1 = Q[off + lane + 32];
    float ss = x0 * x0 + x1 * x1;
    ss += __shfl_xor(ss, 1);
    ss += __shfl_xor(ss, 2);
    ss += __shfl_xor(ss, 4);
    ss += __shfl_xor(ss, 8);
    ss += __shfl_xor(ss, 16);
    float sc = fast_rcp(fmaxf(fast_sqrt(ss), 1e-12f));
    Qh[off + lane]      = (_Float16)(x0 * sc);
    Qh[off + lane + 32] = (_Float16)(x1 * sc);
  }
  {  // K normalize -> f16
    float x0 = K[off + lane], x1 = K[off + lane + 32];
    float ss = x0 * x0 + x1 * x1;
    ss += __shfl_xor(ss, 1);
    ss += __shfl_xor(ss, 2);
    ss += __shfl_xor(ss, 4);
    ss += __shfl_xor(ss, 8);
    ss += __shfl_xor(ss, 16);
    float sc = fast_rcp(fmaxf(fast_sqrt(ss), 1e-12f));
    Kh[off + lane]      = (_Float16)(x0 * sc);
    Kh[off + lane + 32] = (_Float16)(x1 * sc);
  }
  // V -> f16 (unmasked: reference multiplies mask into w, not V)
  Vh[off + lane]      = (_Float16)V[off + lane];
  Vh[off + lane + 32] = (_Float16)V[off + lane + 32];

  // depthwise conv over seq axis of V*mask (correlation, pad 4)
  float a0 = 0.f, a1 = 0.f;
#pragma unroll
  for (int k = 0; k < kKer; ++k) {
    int t = s + k - (kKer / 2);
    if ((unsigned)t < (unsigned)kS) {
      float w  = convw[h * kKer + k] * mask[b * kS + t];
      size_t o = ((size_t)bh * kS + t) * kD;
      a0 += w * V[o + lane];
      a1 += w * V[o + lane + 32];
    }
  }
  convV[off + lane]      = a0;
  convV[off + lane + 32] = a1;
}

// ---------------------------------------------------------------------------
// Fused YOSO attention: per block = 64 query rows of one (b,h); 4 waves,
// each wave owns a 16-row stripe. Stream key tiles of 32 through LDS.
// ---------------------------------------------------------------------------
__global__ __launch_bounds__(128) void yoso_attn(
    const unsigned short* __restrict__ QhU, const unsigned short* __restrict__ KhU,
    const unsigned short* __restrict__ VhU, const float* __restrict__ mask,
    const float* __restrict__ convV, float* __restrict__ out) {
  const _Float16* Qh = reinterpret_cast<const _Float16*>(QhU);
  const _Float16* Kh = reinterpret_cast<const _Float16*>(KhU);
  const _Float16* Vh = reinterpret_cast<const _Float16*>(VhU);

  const int bh   = blockIdx.y;  // 0..B*H-1
  const int b    = bh / kH;
  const int tid  = threadIdx.x;
  const int wave = tid >> 5;
  const int lane = tid & 31;
  const int lmod = lane & 15;
  const int lhi  = lane >> 4;

  __shared__ __align__(16) _Float16 ldsK[32 * 64];    // [key][d]  4 KB
  __shared__ __align__(16) _Float16 ldsVt[64 * 32];   // [d][key]  4 KB
  __shared__ __align__(16) _Float16 ldsW[4][16 * 32]; // per-wave W scratch
  __shared__ float ldsM[32];

  const size_t base = (size_t)bh * kS * kD;
  const int qrow    = blockIdx.x * 64 + wave * 16 + lmod;  // A-layout: M = lane%16

  // A-fragments of Qn (16x64 split into two 16x32 K-chunks), ISA A-layout:
  // k = (v/4)*16 + (v%4)*2 + h + 8*(lane/16)  -> consecutive half pairs per VGPR
  Frag aq[2];
#pragma unroll
  for (int f = 0; f < 2; ++f)
#pragma unroll
    for (int v = 0; v < 8; ++v) {
      int d = 32 * f + (v >> 2) * 16 + (v & 3) * 2 + lhi * 8;
      aq[f].u[v] =
          *reinterpret_cast<const uint32_t*>(Qh + base + (size_t)qrow * kD + d);
    }

  v8f acc[4] = {};  // O[16x64]: 4 N-chunks of 16

  for (int kt = 0; kt < kS / 32; ++kt) {
    const int kbase = kt * 32;
    __syncthreads();  // previous tile fully consumed

    {  // stage K tile [32 keys][64 d], coalesced 16B copies
      const uint4* src =
          reinterpret_cast<const uint4*>(Kh + base + (size_t)kbase * kD);
      uint4* dst     = reinterpret_cast<uint4*>(ldsK);
      dst[tid]       = src[tid];
      dst[tid + 128] = src[tid + 128];
    }
#pragma unroll
    for (int i = 0; i < 16; ++i) {  // stage V tile transposed [d][key]
      int idx = i * 128 + tid;
      int key = idx >> 6;
      int d   = idx & 63;
      ldsVt[d * 32 + key] = Vh[base + (size_t)(kbase + key) * kD + d];
    }
    if (tid < 32) ldsM[tid] = mask[b * kS + kbase + tid];
    __syncthreads();

    if (kt + 1 < kS / 32) {  // gfx1250 global_prefetch of next tile
      __builtin_prefetch(Kh + base + (size_t)(kbase + 32) * kD, 0, 3);
      __builtin_prefetch(Vh + base + (size_t)(kbase + 32) * kD, 0, 3);
    }

    // ---- GEMM1: S[16x32] = Qn(16x64) x Kn^T(64x32), two N-subtiles ----
#pragma unroll
    for (int nsub = 0; nsub < 2; ++nsub) {
      v8f sc = {};
#pragma unroll
      for (int f = 0; f < 2; ++f) {
        Frag bk;  // B-layout: n = lane%16, k = (v/4)*16 + (v%4)*2 + h + 8*(lane/16) (k == d)
#pragma unroll
        for (int v = 0; v < 8; ++v) {
          int d = f * 32 + (v >> 2) * 16 + (v & 3) * 2 + lhi * 8;
          bk.u[v] = *reinterpret_cast<const uint32_t*>(
              &ldsK[(nsub * 16 + lmod) * 64 + d]);
        }
        sc = wmma_f16(aq[f].h, bk.h, sc);
      }
      // elementwise: clip, (1 - acos/pi)^8, key mask; D-layout M = v + 8*lhi
      float mk = ldsM[nsub * 16 + lmod];
#pragma unroll
      for (int v = 0; v < 8; ++v) {
        float x  = sc[v];
        x        = fminf(fmaxf(x, -0.999999f), 0.999999f);
        float t  = one_minus_acos_over_pi(x);
        float t2 = t * t, t4 = t2 * t2;
        ldsW[wave][(v + lhi * 8) * 32 + nsub * 16 + lmod] =
            (_Float16)(t4 * t4 * mk);
      }
    }

    // Re-gather W as an A-fragment (same-wave LDS RAW is in-order)
    Frag aw;
#pragma unroll
    for (int v = 0; v < 8; ++v) {
      int k   = (v >> 2) * 16 + (v & 3) * 2 + lhi * 8;
      aw.u[v] = *reinterpret_cast<const uint32_t*>(&ldsW[wave][lmod * 32 + k]);
    }

    // ---- GEMM2: O[16x64] += W(16x32) x V(32x64), four N-chunks ----
#pragma unroll
    for (int nc = 0; nc < 4; ++nc) {
      Frag bv;  // B-layout over transposed V tile: pairs contiguous in key
#pragma unroll
      for (int v = 0; v < 8; ++v) {
        int k   = (v >> 2) * 16 + (v & 3) * 2 + lhi * 8;
        bv.u[v] = *reinterpret_cast<const uint32_t*>(
            &ldsVt[(nc * 16 + lmod) * 32 + k]);
      }
      acc[nc] = wmma_f16(aw.h, bv.h, acc[nc]);
    }
  }

  // ---- epilogue: query mask, row L2-normalize, + conv_V ----
#pragma unroll
  for (int v = 0; v < 8; ++v) {
    int r    = blockIdx.x * 64 + wave * 16 + v + lhi * 8;  // D-layout row
    float mq = mask[b * kS + r];
    float ss = 0.f;
#pragma unroll
    for (int nc = 0; nc < 4; ++nc) {
      float x    = acc[nc][v] * mq;
      acc[nc][v] = x;
      ss += x * x;
    }
    // reduce across the 16 lanes holding this row's 64 columns
    ss += __shfl_xor(ss, 1);
    ss += __shfl_xor(ss, 2);
    ss += __shfl_xor(ss, 4);
    ss += __shfl_xor(ss, 8);
    float inv = fast_rcp(fmaxf(fast_sqrt(ss), 1e-12f));
    size_t o  = base + (size_t)r * kD;
#pragma unroll
    for (int nc = 0; nc < 4; ++nc) {
      int d      = nc * 16 + lmod;
      out[o + d] = acc[nc][v] * inv + convV[o + d];
    }
  }
}

// ---------------------------------------------------------------------------
extern "C" void kernel_launch(void* const* d_in, const int* in_sizes, int n_in,
                              void* d_out, int out_size, void* d_ws,
                              size_t ws_size, hipStream_t stream) {
  (void)in_sizes; (void)n_in; (void)out_size; (void)ws_size;
  const float* Q    = (const float*)d_in[0];
  const float* K    = (const float*)d_in[1];
  const float* V    = (const float*)d_in[2];
  const float* mask = (const float*)d_in[3];
  const float* cw   = (const float*)d_in[4];
  float* out        = (float*)d_out;

  const size_t N    = (size_t)kB * kH * kS * kD;  // 3,145,728 elements
  unsigned short* Qh = (unsigned short*)d_ws;     // f16  6.29 MB
  unsigned short* Kh = Qh + N;                    // f16  6.29 MB
  unsigned short* Vh = Kh + N;                    // f16  6.29 MB
  float* convV       = (float*)(Vh + N);          // f32 12.58 MB  (~31.5 MB total)

  yoso_prep<<<(kB * kH * kS) / 8, 256, 0, stream>>>(Q, K, V, mask, cw, Qh, Kh,
                                                    Vh, convV);
  dim3 grid(kS / 64, kB * kH);
  yoso_attn<<<grid, 128, 0, stream>>>(Qh, Kh, Vh, mask, convV, out);
}
